// Pre_Block_27015344292114
// MI455X (gfx1250) — matrix operand order repaired and checked by reading, not compiled, not run on hardware
//
#include <hip/hip_runtime.h>
#include <hip/hip_bf16.h>
#include <cstdint>

// ---------------- problem constants ----------------
// B=32, C=512, SEQ=2048, K=4, L=512, M = B*L = B*C = 16384
#define CB_ 32
#define CC_ 512
#define CSEQ_ 2048
#define CL_ 512
#define CM_ 16384

typedef __attribute__((ext_vector_type(16))) __bf16 v16bf;
typedef __attribute__((ext_vector_type(8)))  __bf16 v8bf;
typedef __attribute__((ext_vector_type(8)))  float  v8f;

// ---------------- workspace layout (bytes) ----------------
// region0 (64MB) : im2col patches, recycled after conv GEMM for G / trendT / h
static constexpr size_t OFF_ACONV = 0;                         // bf16 [16384][2048]  64MB
static constexpr size_t OFF_G     = 0;                         // f32  [16384][512]   32MB (reuse)
static constexpr size_t OFF_TRT   = 33554432;                  // bf16 [16384][512]   16MB (reuse)
static constexpr size_t OFF_H     = 50331648;                  // bf16 [16384][512]   16MB (reuse)
// region1
static constexpr size_t OFF_WB    = 67108864;                  // bf16 [512][2048]     2MB
static constexpr size_t OFF_CBB   = OFF_WB  + 2097152;         // bf16 [512][512]
static constexpr size_t OFF_W1    = OFF_CBB + 524288;          // bf16 [512][512]
static constexpr size_t OFF_W2    = OFF_W1  + 524288;          // bf16 [512][512]
static constexpr size_t OFF_CN    = OFF_W2  + 524288;          // f32  [512]
static constexpr size_t OFF_IDX   = OFF_CN  + 4096;            // i32  [16384]
static constexpr size_t OFF_XDE   = OFF_IDX + 65536;           // f32  [B][C][L]      32MB
static constexpr size_t OFF_XDEB  = OFF_XDE + 33554432;        // bf16 [B][C][L]      16MB

// ---------------- CDNA5 async global->LDS helpers ----------------
// GLOBAL_LOAD_ASYNC_TO_LDS_B128: per-lane 16B DMA, LDS dest from VGPR, tracked by ASYNCcnt.
// INST_OFFSET is added to BOTH the LDS and the global address (ISA 08_async_tensor §4.4),
// so the offset:16 form covers the second 16B chunk of a 32B row segment for free.
__device__ __forceinline__ void async_ld_b128(unsigned lds_off, const void* g) {
  asm volatile("global_load_async_to_lds_b128 %0, %1, off"
               :: "v"(lds_off), "v"(g) : "memory");
}
__device__ __forceinline__ void async_ld_b128_o16(unsigned lds_off, const void* g) {
  asm volatile("global_load_async_to_lds_b128 %0, %1, off offset:16"
               :: "v"(lds_off), "v"(g) : "memory");
}
__device__ __forceinline__ void wait_async0() {
#if __has_builtin(__builtin_amdgcn_s_wait_asynccnt)
  __builtin_amdgcn_s_wait_asynccnt(0);
#else
  asm volatile("s_wait_asynccnt 0x0" ::: "memory");
#endif
}

// ---------------- tiny pointwise kernels ----------------
__global__ __launch_bounds__(256) void f32_to_bf16(const float* __restrict__ in,
                                                   __bf16* __restrict__ out, int n) {
  int i = blockIdx.x * 256 + threadIdx.x;
  if (i < n) out[i] = (__bf16)in[i];
}

// cn[k] = sum_c codebook[k][c]^2
__global__ __launch_bounds__(256) void codebook_norms(const float* __restrict__ cb,
                                                      float* __restrict__ cn) {
  int k = blockIdx.x * 256 + threadIdx.x;
  if (k < CL_) {
    float s = 0.f;
    for (int i = 0; i < CC_; ++i) { float v = cb[(size_t)k * CC_ + i]; s += v * v; }
    cn[k] = s;
  }
}

// im2col: Ap[(b*L+l)][i*4+k] = x[b][i][4l+k], converted to bf16
__global__ __launch_bounds__(256) void pack_patches(const float* __restrict__ x,
                                                    __bf16* __restrict__ Ap) {
  int tid = blockIdx.x * 256 + threadIdx.x;          // 2M threads, 16 elems each
  int row = tid >> 7;                                // (b*L + l)
  int seg = tid & 127;                               // 16-col segment within 2048
  int b = row >> 9, l = row & 511;
  int i0 = seg << 2;
  alignas(16) __bf16 vals[16];
#pragma unroll
  for (int ii = 0; ii < 4; ++ii) {
    const float4 v = *(const float4*)&x[(((size_t)b * CC_ + i0 + ii) * CSEQ_) + 4 * l];
    vals[ii * 4 + 0] = (__bf16)v.x; vals[ii * 4 + 1] = (__bf16)v.y;
    vals[ii * 4 + 2] = (__bf16)v.z; vals[ii * 4 + 3] = (__bf16)v.w;
  }
  size_t o = (size_t)row * 2048 + seg * 16;
  *(uint4*)&Ap[o]     = *(uint4*)&vals[0];
  *(uint4*)&Ap[o + 8] = *(uint4*)&vals[8];
}

// argmin_k( cn[k] - 2*G[row][k] )  (xn per-row constant dropped), wave per row
__global__ __launch_bounds__(256) void argmin_rows(const float* __restrict__ G,
                                                   const float* __restrict__ cn,
                                                   int* __restrict__ idx) {
  int row  = blockIdx.x * 8 + (threadIdx.x >> 5);
  int lane = threadIdx.x & 31;
  float best = 3.4e38f; int bi = 0;
  for (int k = lane; k < CL_; k += 32) {
    float d = cn[k] - 2.0f * G[(size_t)row * CL_ + k];
    if (d < best) { best = d; bi = k; }
  }
#pragma unroll
  for (int off = 16; off > 0; off >>= 1) {
    float ov = __shfl_xor(best, off, 32);
    int   oi = __shfl_xor(bi, off, 32);
    if (ov < best || (ov == best && oi < bi)) { best = ov; bi = oi; }
  }
  if (lane == 0) idx[row] = bi;
}

// trendT[(b*L+l)][c] = x_de[b][c][l] - codebook[idx[b,c]][l]  (LDS tile transpose)
__global__ __launch_bounds__(256) void trend_transpose(const float* __restrict__ xde,
                                                       const float* __restrict__ cbF,
                                                       const int* __restrict__ idx,
                                                       __bf16* __restrict__ trT) {
  __shared__ __bf16 tile[64 * 65];
  int b  = blockIdx.z;
  int c0 = blockIdx.y * 64;
  int l0 = blockIdx.x * 64;
  int t  = threadIdx.x;
  // phase 1: coalesced reads along l, write tile[c][l]
  int cl = t >> 2;
  int ls = (t & 3) * 16;
  int ci = idx[b * CC_ + c0 + cl];
  const float* xr = xde + ((size_t)b * CC_ + c0 + cl) * CL_ + l0 + ls;
  const float* qr = cbF + (size_t)ci * CL_ + l0 + ls;
#pragma unroll
  for (int j = 0; j < 16; ++j)
    tile[cl * 65 + ls + j] = (__bf16)(xr[j] - qr[j]);
  __syncthreads();
  // phase 2: coalesced writes along c
  int ll = t >> 2;
  int cs = (t & 3) * 16;
  alignas(16) __bf16 vals[16];
#pragma unroll
  for (int j = 0; j < 16; ++j) vals[j] = tile[(cs + j) * 65 + ll];
  size_t base = ((size_t)b * CL_ + l0 + ll) * CC_ + c0 + cs;
  *(uint4*)&trT[base]     = *(uint4*)&vals[0];
  *(uint4*)&trT[base + 8] = *(uint4*)&vals[8];
}

// ---------------- WMMA GEMM: D = A[M,K] * Bt[N,K]^T, bf16 in / f32 acc ----------------
// block = 8 waves, tile 128(M) x 64(N), K-slice 32; wave tile 32x32 = 2x2 WMMA.
// K slices double-buffered in LDS via GLOBAL_LOAD_ASYNC_TO_LDS_B128 (ASYNCcnt):
// issue slice k+1, compute slice k's 4 WMMAs, s_wait_asynccnt 0, one barrier/iter.
// EPI: 0=conv (bias, write xde f32+bf16 at [B,C,L]); 1=plain f32 [M,N];
//      2=bias+relu -> bf16 [M,N]; 3=bias + codebook gather + transposed write to out
template <int EPI>
__global__ __launch_bounds__(256) void gemm_bf16(
    const __bf16* __restrict__ A, const __bf16* __restrict__ Bt,
    int Kn, int Nn,
    const float* __restrict__ bias,
    float* __restrict__ outF, __bf16* __restrict__ outBF,
    const int* __restrict__ idx, const float* __restrict__ cbF) {
  __shared__ __bf16 As[2][128 * 40];   // 32 K + 8 pad per row, double buffered
  __shared__ __bf16 Bs[2][64 * 40];

  const int t = threadIdx.x;
  const int lane = t & 31, w = t >> 5;
  const int wm = w >> 1, wn = w & 1;
  const int lrow = lane & 15, hi = lane >> 4;
  const int bm = blockIdx.y * 128, bn = blockIdx.x * 64;

  const v8f vz = {0.f, 0.f, 0.f, 0.f, 0.f, 0.f, 0.f, 0.f};
  v8f acc[2][2] = {{vz, vz}, {vz, vz}};

  // staging: A row 0..127 / 16-col chunk; B row 0..63 / 8-col chunk (16B per lane each)
  const int ar = t >> 1, ac = (t & 1) * 16;
  const int br = t >> 2, bc = (t & 3) * 8;
  const unsigned laA = (unsigned)(uintptr_t)&As[0][ar * 40 + ac];
  const unsigned laB = (unsigned)(uintptr_t)&Bs[0][br * 40 + bc];
  const unsigned bufA = (unsigned)(128 * 40 * sizeof(__bf16));   // 10240B
  const unsigned bufB = (unsigned)(64 * 40 * sizeof(__bf16));    //  5120B
  const __bf16* gA = A + (size_t)(bm + ar) * Kn + ac;
  const __bf16* gB = Bt + (size_t)(bn + br) * Kn + bc;

  const int nk = Kn >> 5;
  // prologue: DMA slice 0 into buffer 0
  async_ld_b128(laA, gA);
  async_ld_b128_o16(laA, gA);
  async_ld_b128(laB, gB);
  wait_async0();
  __syncthreads();

  for (int i = 0; i < nk; ++i) {
    if (i + 1 < nk) {                       // issue next slice's DMA (ping-pong)
      const unsigned p = (unsigned)((i + 1) & 1);
      const __bf16* ga = gA + (size_t)(i + 1) * 32;
      async_ld_b128(laA + p * bufA, ga);
      async_ld_b128_o16(laA + p * bufA, ga);
      async_ld_b128(laB + p * bufB, gB + (size_t)(i + 1) * 32);
    }

    const __bf16* Ab = As[i & 1];
    const __bf16* Bb = Bs[i & 1];
    v16bf af[2], bfv[2];
#pragma unroll
    for (int mt = 0; mt < 2; ++mt) {
      // A 16x32 bf16 as loaded here (same k order as B): lane holds row M=lane%16; K = (lane/16)*16 .. +15 contiguous
      const __bf16* p = &Ab[(wm * 32 + mt * 16 + lrow) * 40];
      v8bf lo = *(const v8bf*)(p + hi * 16);
      v8bf hh = *(const v8bf*)(p + hi * 16 + 8);
      af[mt] = __builtin_shufflevector(lo, hh, 0, 1, 2, 3, 4, 5, 6, 7,
                                       8, 9, 10, 11, 12, 13, 14, 15);
    }
#pragma unroll
    for (int nt = 0; nt < 2; ++nt) {
      // B 32x16 bf16 as loaded here (same k order as A): lane holds col N=lane%16; K = (lane/16)*16 .. +15 contiguous
      const __bf16* p = &Bb[(wn * 32 + nt * 16 + lrow) * 40 + hi * 16];
      v8bf lo = *(const v8bf*)(p);
      v8bf hh = *(const v8bf*)(p + 8);
      bfv[nt] = __builtin_shufflevector(lo, hh, 0, 1, 2, 3, 4, 5, 6, 7,
                                        8, 9, 10, 11, 12, 13, 14, 15);
    }
#pragma unroll
    for (int mt = 0; mt < 2; ++mt)
#pragma unroll
      for (int nt = 0; nt < 2; ++nt)
        acc[mt][nt] = __builtin_amdgcn_wmma_f32_16x16x32_bf16(
            false, af[mt], false, bfv[nt], (short)0, acc[mt][nt], false, false);

    if (i + 1 < nk) { wait_async0(); __syncthreads(); }
  }

  // epilogue: C/D layout -> lane<16: (M=v, N=lane); lane>=16: (M=8+v, N=lane-16)
#pragma unroll
  for (int mt = 0; mt < 2; ++mt) {
#pragma unroll
    for (int nt = 0; nt < 2; ++nt) {
#pragma unroll
      for (int v = 0; v < 8; ++v) {
        int r = bm + wm * 32 + mt * 16 + v + 8 * hi;
        int c = bn + wn * 32 + nt * 16 + lrow;
        float val = acc[mt][nt][v];
        if (EPI == 0) {                       // conv: r=(b,l), c=channel o
          int b = r >> 9, l = r & 511;
          float o = val + bias[c];
          size_t p = ((size_t)b * CC_ + c) * CL_ + l;
          outF[p]  = o;
          outBF[p] = (__bf16)o;
        } else if (EPI == 1) {                // distance GEMM: raw G
          outF[(size_t)r * Nn + c] = val;
        } else if (EPI == 2) {                // MLP1: bias + relu -> bf16
          float o = val + bias[c];
          o = o > 0.f ? o : 0.f;
          outBF[(size_t)r * Nn + c] = (__bf16)o;
        } else {                              // MLP2: bias + quantized, transposed write
          int b = r >> 9, l = r & 511;
          float q = cbF[(size_t)idx[b * CC_ + c] * CL_ + l];
          outF[((size_t)b * CC_ + c) * CL_ + l] = val + bias[c] + q;
        }
      }
    }
  }
}

// ---------------- host launcher ----------------
extern "C" void kernel_launch(void* const* d_in, const int* in_sizes, int n_in,
                              void* d_out, int out_size, void* d_ws, size_t ws_size,
                              hipStream_t stream) {
  const float* x        = (const float*)d_in[0];
  const float* conv_w   = (const float*)d_in[1];
  const float* conv_b   = (const float*)d_in[2];
  const float* codebook = (const float*)d_in[3];
  const float* w1       = (const float*)d_in[4];
  const float* b1       = (const float*)d_in[5];
  const float* w2       = (const float*)d_in[6];
  const float* b2       = (const float*)d_in[7];
  char* ws = (char*)d_ws;

  __bf16* Aconv = (__bf16*)(ws + OFF_ACONV);
  float*  G     = (float*) (ws + OFF_G);
  __bf16* trT   = (__bf16*)(ws + OFF_TRT);
  __bf16* Hbf   = (__bf16*)(ws + OFF_H);
  __bf16* Wb    = (__bf16*)(ws + OFF_WB);
  __bf16* cbB   = (__bf16*)(ws + OFF_CBB);
  __bf16* w1b   = (__bf16*)(ws + OFF_W1);
  __bf16* w2b   = (__bf16*)(ws + OFF_W2);
  float*  cn    = (float*) (ws + OFF_CN);
  int*    idx   = (int*)   (ws + OFF_IDX);
  float*  xde   = (float*) (ws + OFF_XDE);
  __bf16* xdeB  = (__bf16*)(ws + OFF_XDEB);
  float*  out   = (float*)d_out;

  // weight conversions (bf16) + codebook norms
  f32_to_bf16<<<(CC_ * CSEQ_ + 255) / 256, 256, 0, stream>>>(conv_w, Wb, CC_ * CSEQ_);
  f32_to_bf16<<<(CL_ * CC_ + 255) / 256, 256, 0, stream>>>(codebook, cbB, CL_ * CC_);
  f32_to_bf16<<<(CL_ * CL_ + 255) / 256, 256, 0, stream>>>(w1, w1b, CL_ * CL_);
  f32_to_bf16<<<(CL_ * CL_ + 255) / 256, 256, 0, stream>>>(w2, w2b, CL_ * CL_);
  codebook_norms<<<2, 256, 0, stream>>>(codebook, cn);

  // im2col
  pack_patches<<<8192, 256, 0, stream>>>(x, Aconv);

  dim3 g1(CC_ / 64, CM_ / 128);  // (8, 128)

  // 1) conv GEMM -> x_de (f32 + bf16 mirror), laid out [B,C,L]
  gemm_bf16<0><<<g1, 256, 0, stream>>>(Aconv, Wb, CSEQ_, CC_, conv_b, xde, xdeB, nullptr, nullptr);
  // 2) distance GEMM -> G = x_de . codebook^T   (reuses Aconv region)
  gemm_bf16<1><<<g1, 256, 0, stream>>>(xdeB, cbB, CC_, CL_, nullptr, G, nullptr, nullptr, nullptr);
  // 3) argmin over codebook
  argmin_rows<<<CM_ / 8, 256, 0, stream>>>(G, cn, idx);
  // 4) trend = x_de - codebook[idx], transposed to [(b,l), c] bf16
  trend_transpose<<<dim3(CL_ / 64, CC_ / 64, CB_), 256, 0, stream>>>(xde, codebook, idx, trT);
  // 5) MLP layer 1: relu(trend . w1^T + b1) -> bf16
  gemm_bf16<2><<<g1, 256, 0, stream>>>(trT, w1b, CC_, CL_, b1, nullptr, Hbf, nullptr, nullptr);
  // 6) MLP layer 2 + quantized residual, transposed into final [B,C,L] output
  gemm_bf16<3><<<g1, 256, 0, stream>>>(Hbf, w2b, CL_, CL_, b2, out, nullptr, idx, codebook);
}
